// Block_70531952935537
// MI455X (gfx1250) — matrix operand order repaired and checked by reading, not compiled, not run on hardware
//
#include <hip/hip_runtime.h>
#include <hip/hip_bf16.h>
#include <cmath>
#include <cstdint>

typedef _Float16 f16;
typedef __attribute__((ext_vector_type(16))) _Float16 v16h;
typedef __attribute__((ext_vector_type(8)))  _Float16 v8h;
typedef __attribute__((ext_vector_type(4)))  _Float16 v4h;
typedef __attribute__((ext_vector_type(8)))  float    v8f;
typedef __attribute__((ext_vector_type(4)))  float    v4f;

#define LN_EPS 1e-5f
#define ATTN_SCALE 0.03125f   // DIM^-0.5 = 1/32

// ---------------------------------------------------------------------------
// WMMA helper (CDNA5 V_WMMA_F32_16X16X32_F16, wave32)
// ---------------------------------------------------------------------------
__device__ __forceinline__ v8f wmma_f16(v16h a, v16h b, v8f c) {
  return __builtin_amdgcn_wmma_f32_16x16x32_f16(false, a, false, b, (short)0, c,
                                                false, false);
}

// Load 8 consecutive elements (f32 or f16 source) as 8 halves, vectorized.
__device__ __forceinline__ v8h load8(const float* p) {
  const v4f a = *(const v4f*)p;
  const v4f b = *(const v4f*)(p + 4);
  v8h o;
#pragma unroll
  for (int i = 0; i < 4; ++i) { o[i] = (f16)a[i]; o[4 + i] = (f16)b[i]; }
  return o;
}
__device__ __forceinline__ v8h load8(const f16* p) { return *(const v8h*)p; }

// Async global->LDS copy of 16 bytes per lane (CDNA5, tracked by ASYNCcnt).
__device__ __forceinline__ void async_copy_b128(uint32_t lds_byte_off,
                                                const void* gaddr) {
  asm volatile("global_load_async_to_lds_b128 %0, %1, off"
               :: "v"(lds_byte_off), "v"((unsigned long long)(uintptr_t)gaddr)
               : "memory");
}
__device__ __forceinline__ void wait_asynccnt0() {
#if __has_builtin(__builtin_amdgcn_s_wait_asynccnt)
  __builtin_amdgcn_s_wait_asynccnt(0);
#else
  asm volatile("s_wait_asynccnt 0x0" ::: "memory");
#endif
}

// ---------------------------------------------------------------------------
// LayerNorm: one block per row of 1024, fp32 in/out, float4 I/O
// ---------------------------------------------------------------------------
__global__ __launch_bounds__(256) void ln_kernel(const float* __restrict__ x,
                                                 const float* __restrict__ g,
                                                 const float* __restrict__ be,
                                                 float* __restrict__ out) {
  __shared__ float sred[256];
  __shared__ float stat[2];
  const int tid = threadIdx.x;
  const size_t base = (size_t)blockIdx.x * 1024;

  const v4f v = *(const v4f*)&x[base + tid * 4];
  float s1 = 0.f, s2 = 0.f;
#pragma unroll
  for (int i = 0; i < 4; ++i) { s1 += v[i]; s2 += v[i] * v[i]; }
  sred[tid] = s1; __syncthreads();
  for (int s = 128; s > 0; s >>= 1) { if (tid < s) sred[tid] += sred[tid + s]; __syncthreads(); }
  if (tid == 0) stat[0] = sred[0] * (1.f / 1024.f);
  __syncthreads();
  sred[tid] = s2; __syncthreads();
  for (int s = 128; s > 0; s >>= 1) { if (tid < s) sred[tid] += sred[tid + s]; __syncthreads(); }
  if (tid == 0) stat[1] = sred[0] * (1.f / 1024.f);
  __syncthreads();

  const float mu   = stat[0];
  const float rstd = rsqrtf(stat[1] - mu * mu + LN_EPS);
  const v4f gg = *(const v4f*)&g[tid * 4];
  const v4f bb = *(const v4f*)&be[tid * 4];
  v4f o;
#pragma unroll
  for (int i = 0; i < 4; ++i) o[i] = (v[i] - mu) * rstd * gg[i] + bb[i];
  *(v4f*)&out[base + tid * 4] = o;
}

// ---------------------------------------------------------------------------
// WMMA GEMM: out = epilogue(A[MxK] @ W[KxN] + bias)
// Block tile 128x128, K-step 32; 8 waves each own 32x64 (2x4 16x16 frags).
// A tile stored in LDS in *fragment order*: Asw[frag][lane][16 halves]
//   element (row m, k): frag=row>>4, lane=((k>>3)&1)<<4 | (m&15),
//                       j = ((k>>4)&1)*8 + (k&7)
// EPI: 0 f32 store | 1 f32 + residual | 2 exact GELU -> f16 | 3 f16 store
// ---------------------------------------------------------------------------
template <typename TA, int EPI>
__global__ __launch_bounds__(256) void gemm_wmma_kernel(
    const TA* __restrict__ A, const float* __restrict__ W,
    const float* __restrict__ bias, const float* __restrict__ resid,
    float* __restrict__ outF, f16* __restrict__ outH, int M, int N, int K) {
  __shared__ f16 Asw[8][32][16];   // 8 KB, fragment-ordered A tile (128x32)
  __shared__ f16 Ws[32][144];      // 9.2 KB, row-major W tile (32x128, 288B rows)

  const int tid  = threadIdx.x;
  const int lane = tid & 31;
  const int wave = tid >> 5;
  const int wm   = wave >> 1;      // 0..3 : 32-row strip
  const int wn   = wave & 1;       // 0..1 : 64-col strip
  const int m0   = blockIdx.y * 128;
  const int n0   = blockIdx.x * 128;

  v8f acc[2][4];
#pragma unroll
  for (int i = 0; i < 2; ++i)
#pragma unroll
    for (int j = 0; j < 4; ++j)
#pragma unroll
      for (int r = 0; r < 8; ++r) acc[i][j][r] = 0.f;

  for (int k0 = 0; k0 < K; k0 += 32) {
    // stage A: 512 chunks of 8 halves, swizzled into fragment order
#pragma unroll
    for (int t = 0; t < 2; ++t) {
      const int c   = tid + t * 256;
      const int row = c >> 2;
      const int kq  = (c & 3) * 8;
      const v8h val = load8(&A[(size_t)(m0 + row) * K + k0 + kq]);
      const int fi  = row >> 4;
      const int ln  = (((kq >> 3) & 1) << 4) | (row & 15);
      const int j0  = ((kq >> 4) & 1) * 8;
      *(v8h*)&Asw[fi][ln][j0] = val;
    }
    // stage W: 1024 groups of 4 floats -> 4 halves (zero-pad N edge)
#pragma unroll
    for (int t = 0; t < 4; ++t) {
      const int g  = tid + t * 256;
      const int r  = g >> 5;
      const int c0 = (g & 31) * 4;
      v4h wv;
      if (n0 + c0 + 3 < N) {
        const v4f w4 = *(const v4f*)&W[(size_t)(k0 + r) * N + n0 + c0];
#pragma unroll
        for (int i = 0; i < 4; ++i) wv[i] = (f16)w4[i];
      } else {
#pragma unroll
        for (int i = 0; i < 4; ++i)
          wv[i] = (f16)((n0 + c0 + i < N) ? W[(size_t)(k0 + r) * N + n0 + c0 + i] : 0.f);
      }
      *(v4h*)&Ws[r][c0] = wv;
    }
    if (k0 + 32 < K)
      __builtin_prefetch(&A[(size_t)(m0 + (tid >> 1)) * K + k0 + 32], 0, 1);
    __syncthreads();

    const v16h a0 = *(const v16h*)&Asw[wm * 2][lane][0];
    const v16h a1 = *(const v16h*)&Asw[wm * 2 + 1][lane][0];
#pragma unroll
    for (int sub = 0; sub < 4; ++sub) {
      v16h bb;
#pragma unroll
      for (int s = 0; s < 16; ++s)
        bb[s] = Ws[((s >> 3) << 4) + ((lane >> 4) << 3) + (s & 7)][wn * 64 + sub * 16 + (lane & 15)];
      acc[0][sub] = wmma_f16(a0, bb, acc[0][sub]);
      acc[1][sub] = wmma_f16(a1, bb, acc[1][sub]);
    }
    __syncthreads();
  }

  // epilogue: C/D layout m = r + 8*(lane>=16), n = lane&15
  const int mofs = ((lane >> 4) << 3);
  const int nofs = (lane & 15);
#pragma unroll
  for (int i = 0; i < 2; ++i)
#pragma unroll
    for (int sub = 0; sub < 4; ++sub)
#pragma unroll
      for (int r = 0; r < 8; ++r) {
        const int m = m0 + wm * 32 + i * 16 + r + mofs;
        const int n = n0 + wn * 64 + sub * 16 + nofs;
        if (m < M && n < N) {
          float v = acc[i][sub][r] + (bias ? bias[n] : 0.f);
          if (EPI == 1) v += resid[(size_t)m * N + n];
          if (EPI == 0 || EPI == 1) {
            outF[(size_t)m * N + n] = v;
          } else if (EPI == 2) {
            outH[(size_t)m * N + n] = (f16)(0.5f * v * (1.f + erff(v * 0.70710678118f)));
          } else {
            outH[(size_t)m * N + n] = (f16)v;
          }
        }
      }
}

// ---------------------------------------------------------------------------
// Flash attention, transposed-score form:  S' = K_tile x Q^T
// grid (N/128, B*H); 8 waves; wave w owns q columns w*16..w*16+15.
// - Q^T staged to LDS once per block (B operand, loop-invariant in regs)
// - K tile staged fragment-ordered (A operand) with b128 ld/st
// - V tile staged row-major via global_load_async_to_lds_b128 (B operand)
// - P fragment built ENTIRELY in registers: S' C-layout slot (kf,r) at a lane
//   is exactly A-layout slot j=kf*8+r for contraction key kf*16+8*hs+r.
// ---------------------------------------------------------------------------
__global__ __launch_bounds__(256) void attn_kernel(const f16* __restrict__ Q,
                                                   const f16* __restrict__ K,
                                                   const f16* __restrict__ V,
                                                   f16* __restrict__ O) {
  __shared__ f16 Qts[64][144];        // Q^T: [d][q], 288B rows -> 18.4 KB
  __shared__ f16 Ksw[2][2][32][16];   // K tile fragment-ordered: 4 KB
  __shared__ f16 Vs[32][80];          // V tile row-major, 160B rows: 5 KB

  const int tid  = threadIdx.x;
  const int lane = tid & 31;
  const int wave = tid >> 5;
  const int b    = blockIdx.y >> 4;
  const int h    = blockIdx.y & 15;
  const size_t base = (size_t)b * 2048 * 1024 + (size_t)h * 64;
  const int q0blk = blockIdx.x * 128;
  const int myhs  = lane >> 4;
  const int qcol  = lane & 15;

  // ---- stage Q^T once (transpose scatter, outside the k-loop) ----
  for (int c = tid; c < 1024; c += 256) {       // 128 q-rows x 8 d-groups
    const int qr = c >> 3;
    const int d0 = (c & 7) * 8;
    const v8h qv = *(const v8h*)&Q[base + (size_t)(q0blk + qr) * 1024 + d0];
#pragma unroll
    for (int i = 0; i < 8; ++i) Qts[d0 + i][qr] = qv[i];
  }
  __syncthreads();

  // loop-invariant B operands: Q^T fragments for this wave's 16 q columns
  v16h bq0, bq1;
#pragma unroll
  for (int s = 0; s < 16; ++s) {
    const int kk = ((s >> 3) << 4) + (myhs << 3) + (s & 7);
    bq0[s] = Qts[kk][wave * 16 + qcol];
    bq1[s] = Qts[32 + kk][wave * 16 + qcol];
  }

  v8f oacc[4];
#pragma unroll
  for (int f = 0; f < 4; ++f)
#pragma unroll
    for (int r = 0; r < 8; ++r) oacc[f][r] = 0.f;
  float mrun = -1e30f, lrun = 0.f;   // per-lane: stats of q column `qcol`

  const int skey = tid & 31;          // staging: key row
  const int sd0  = (tid >> 5) * 8;    // staging: 8-aligned d group

  for (int kt = 0; kt < 2048; kt += 32) {
    // V tile: async global->LDS, 16B per thread
    async_copy_b128((uint32_t)(uintptr_t)&Vs[skey][sd0],
                    &V[base + (size_t)(kt + skey) * 1024 + sd0]);
    // K tile: fragment-ordered staging, one b128 ld + one b128 st per thread
    {
      const v8h kv = *(const v8h*)&K[base + (size_t)(kt + skey) * 1024 + sd0];
      const int kf = skey >> 4, m = skey & 15;
      const int df = sd0 >> 5, dd0 = sd0 & 31;
      const int ln = (((dd0 >> 3) & 1) << 4) | m;
      const int j0 = ((dd0 >> 4) & 1) * 8;
      *(v8h*)&Ksw[kf][df][ln][j0] = kv;
    }
    wait_asynccnt0();
    __syncthreads();

    // S' = K x Q^T  (C: rows=keys, cols=q)
    v8f s[2];
#pragma unroll
    for (int kf = 0; kf < 2; ++kf) {
      const v16h a0 = *(const v16h*)&Ksw[kf][0][lane][0];
      const v16h a1 = *(const v16h*)&Ksw[kf][1][lane][0];
      v8f z;
#pragma unroll
      for (int r = 0; r < 8; ++r) z[r] = 0.f;
      s[kf] = wmma_f16(a0, bq0, z);
      s[kf] = wmma_f16(a1, bq1, s[kf]);
    }

    // online softmax over this lane's q column (16 local keys + lane^16)
    float mx = -1e30f;
#pragma unroll
    for (int kf = 0; kf < 2; ++kf)
#pragma unroll
      for (int r = 0; r < 8; ++r) mx = fmaxf(mx, s[kf][r] * ATTN_SCALE);
    mx = fmaxf(mx, __shfl_xor(mx, 16, 32));
    const float mn    = fmaxf(mrun, mx);
    const float alpha = __expf(mrun - mn);
    mrun = mn;

    v16h ap;                     // P fragment, built in registers
    float rs = 0.f;
#pragma unroll
    for (int kf = 0; kf < 2; ++kf)
#pragma unroll
      for (int r = 0; r < 8; ++r) {
        const float p = __expf(s[kf][r] * ATTN_SCALE - mn);
        rs += p;
        ap[kf * 8 + r] = (f16)p;
      }
    rs += __shfl_xor(rs, 16, 32);
    lrun = lrun * alpha + rs;

    // rescale O (O rows are q = r + 8*myhs; alpha lives at lane q)
    float arow[8];
#pragma unroll
    for (int r = 0; r < 8; ++r) arow[r] = __shfl(alpha, r + 8 * myhs, 32);
#pragma unroll
    for (int f = 0; f < 4; ++f)
#pragma unroll
      for (int r = 0; r < 8; ++r) oacc[f][r] *= arow[r];

    // O += P x V  (B operand: lane = d column, 16 keys per lane in table order)
#pragma unroll
    for (int f = 0; f < 4; ++f) {
      v16h bv;
#pragma unroll
      for (int s = 0; s < 16; ++s)
        bv[s] = Vs[((s >> 3) << 4) + (myhs << 3) + (s & 7)][f * 16 + qcol];
      oacc[f] = wmma_f16(ap, bv, oacc[f]);
    }
    __syncthreads();
  }

  // final normalize + store: O row q = r + 8*myhs, col d = f*16 + qcol
  float lrow[8];
#pragma unroll
  for (int r = 0; r < 8; ++r) lrow[r] = __shfl(lrun, r + 8 * myhs, 32);
#pragma unroll
  for (int f = 0; f < 4; ++f)
#pragma unroll
    for (int r = 0; r < 8; ++r) {
      const int qrow = q0blk + wave * 16 + r + 8 * myhs;
      O[base + (size_t)qrow * 1024 + f * 16 + qcol] = (f16)(oacc[f][r] / lrow[r]);
    }
}

// ---------------------------------------------------------------------------
// Row softmax over 1000 logits -> d_out (fp32)
// ---------------------------------------------------------------------------
__global__ __launch_bounds__(256) void softmax_kernel(const float* __restrict__ logits,
                                                      float* __restrict__ out) {
  __shared__ float sred[256];
  __shared__ float stat;
  const int tid = threadIdx.x;
  const size_t base = (size_t)blockIdx.x * 1000;

  float v[4];
  float mx = -1e30f;
#pragma unroll
  for (int i = 0; i < 4; ++i) {
    const int c = tid + i * 256;
    v[i] = (c < 1000) ? logits[base + c] : -1e30f;
    mx = fmaxf(mx, v[i]);
  }
  sred[tid] = mx; __syncthreads();
  for (int s = 128; s > 0; s >>= 1) { if (tid < s) sred[tid] = fmaxf(sred[tid], sred[tid + s]); __syncthreads(); }
  if (tid == 0) stat = sred[0];
  __syncthreads();
  const float m = stat;

  float sum = 0.f;
#pragma unroll
  for (int i = 0; i < 4; ++i) {
    const int c = tid + i * 256;
    v[i] = (c < 1000) ? __expf(v[i] - m) : 0.f;
    sum += v[i];
  }
  sred[tid] = sum; __syncthreads();
  for (int s = 128; s > 0; s >>= 1) { if (tid < s) sred[tid] += sred[tid + s]; __syncthreads(); }
  if (tid == 0) stat = 1.f / sred[0];
  __syncthreads();
  const float inv = stat;

#pragma unroll
  for (int i = 0; i < 4; ++i) {
    const int c = tid + i * 256;
    if (c < 1000) out[base + c] = v[i] * inv;
  }
}

// ---------------------------------------------------------------------------
// Orchestration
// ---------------------------------------------------------------------------
extern "C" void kernel_launch(void* const* d_in, const int* in_sizes, int n_in,
                              void* d_out, int out_size, void* d_ws, size_t ws_size,
                              hipStream_t stream) {
  (void)in_sizes; (void)n_in; (void)out_size; (void)ws_size;

  const float* x     = (const float*)d_in[0];
  const float* value = (const float*)d_in[1];
  const float* wq    = (const float*)d_in[2];
  const float* wk    = (const float*)d_in[3];
  const float* wv    = (const float*)d_in[4];
  const float* w_ap  = (const float*)d_in[5];
  const float* b_ap  = (const float*)d_in[6];
  const float* g1    = (const float*)d_in[7];
  const float* be1   = (const float*)d_in[8];
  const float* g2    = (const float*)d_in[9];
  const float* be2   = (const float*)d_in[10];
  const float* w1    = (const float*)d_in[11];
  const float* b1    = (const float*)d_in[12];
  const float* w2    = (const float*)d_in[13];
  const float* b2    = (const float*)d_in[14];
  const float* w_out = (const float*)d_in[15];
  const float* b_out = (const float*)d_in[16];

  char* ws = (char*)d_ws;
  const size_t MB = 1u << 20;
  float* xn     = (float*)(ws + 0);          // 32 MB  [dead after q,k]
  f16*   qh     = (f16*)(ws + 32 * MB);      // 16 MB
  f16*   kh     = (f16*)(ws + 48 * MB);      // 16 MB
  f16*   vh     = (f16*)(ws + 64 * MB);      // 16 MB
  f16*   oh     = (f16*)(ws + 80 * MB);      // 16 MB  [dead after attn-proj]
  float* x1     = (float*)(ws + 96 * MB);    // 32 MB
  float* x1n    = (float*)(ws + 0);          // reuse xn
  f16*   hh     = (f16*)(ws + 32 * MB);      // 64 MB, reuse qh..oh
  float* x2     = (float*)(ws + 128 * MB);   // 32 MB
  float* logits = (float*)(ws + 160 * MB);   // ~31.3 MB

  const int M = 4 * 2048;            // 8192 tokens
  const dim3 blk(256);
  const dim3 gN1024(8,  M / 128);    // N=1024 / 128-col tiles
  const dim3 gN4096(32, M / 128);    // N=4096
  const dim3 gN1000(8,  M / 128);    // ceil(1000/128)=8

  ln_kernel<<<M, blk, 0, stream>>>(x, g1, be1, xn);
  gemm_wmma_kernel<float, 3><<<gN1024, blk, 0, stream>>>(xn, wq, nullptr, nullptr, nullptr, qh, M, 1024, 1024);
  gemm_wmma_kernel<float, 3><<<gN1024, blk, 0, stream>>>(xn, wk, nullptr, nullptr, nullptr, kh, M, 1024, 1024);
  gemm_wmma_kernel<float, 3><<<gN1024, blk, 0, stream>>>(value, wv, nullptr, nullptr, nullptr, vh, M, 1024, 1024);
  attn_kernel<<<dim3(16, 64), blk, 0, stream>>>(qh, kh, vh, oh);
  gemm_wmma_kernel<f16, 1><<<gN1024, blk, 0, stream>>>(oh, w_ap, b_ap, x, x1, nullptr, M, 1024, 1024);
  ln_kernel<<<M, blk, 0, stream>>>(x1, g2, be2, x1n);
  gemm_wmma_kernel<float, 2><<<gN4096, blk, 0, stream>>>(x1n, w1, b1, nullptr, nullptr, hh, M, 4096, 1024);
  gemm_wmma_kernel<f16, 1><<<gN1024, blk, 0, stream>>>(hh, w2, b2, x1, x2, nullptr, M, 1024, 4096);
  gemm_wmma_kernel<float, 0><<<gN1000, blk, 0, stream>>>(x2, w_out, b_out, nullptr, logits, nullptr, M, 1000, 1024);
  softmax_kernel<<<M, blk, 0, stream>>>(logits, (float*)d_out);
}
